// DraftAttention_89266600280667
// MI455X (gfx1250) — matrix operand order repaired and checked by reading, not compiled, not run on hardware
//
#include <hip/hip_runtime.h>
#include <math.h>

// ---------------- problem constants ----------------
#define B_    2
#define S_    2048
#define H_    8
#define HKV_  4
#define D_    256
#define IN_   5120            // INPUT_DIM
#define HID_  2560            // HIDDEN
#define MTOT  (B_*S_)         // 4096 rows of activations

typedef __bf16 bf16;
typedef __attribute__((ext_vector_type(16))) __bf16 v16bf;
typedef __attribute__((ext_vector_type(8)))  __bf16 v8bf;
typedef __attribute__((ext_vector_type(8)))  float  v8f;

union BFrag { v16bf v; v8bf h[2]; };

__device__ inline v8f wmma_bf16(v16bf a, v16bf b, v8f c) {
  return __builtin_amdgcn_wmma_f32_16x16x32_bf16(false, a, false, b, (short)0, c,
                                                 false, false);
}

// ---------------------------------------------------------------------------
// CDNA5 async global->LDS 16B copy (ASYNCcnt path), with sync fallback.
// Builtin signature (probe-derived): (v4i AS1* src, v4i AS3* dst, Ii, Ii).
// LDS offset = low 32 bits of the generic shared address (aperture mapping).
// ---------------------------------------------------------------------------
#if defined(__HIP_DEVICE_COMPILE__) && __has_builtin(__builtin_amdgcn_global_load_async_to_lds_b128)
#define HAS_ASYNC_LDS 1
typedef int vi4 __attribute__((vector_size(16)));
typedef __attribute__((address_space(1))) vi4 gvi4;   // global int4
typedef __attribute__((address_space(3))) vi4 lvi4;   // LDS int4
__device__ inline void cp16_async(void* lds, const void* glb) {
  __builtin_amdgcn_global_load_async_to_lds_b128(
      (gvi4*)(unsigned long long)(size_t)glb,
      (lvi4*)(unsigned)(size_t)lds,
      0, 0);
}
#else
#define HAS_ASYNC_LDS 0
__device__ inline void cp16_async(void* lds, const void* glb) {
  *(uint4*)lds = *(const uint4*)glb;
}
#endif

__device__ inline void wait_async0() {
#if defined(__HIP_DEVICE_COMPILE__) && __has_builtin(__builtin_amdgcn_s_wait_asynccnt)
  __builtin_amdgcn_s_wait_asynccnt(0);
#elif defined(__HIP_DEVICE_COMPILE__)
  asm volatile("s_wait_asynccnt 0x0" ::: "memory");
#endif
}

// ---------------------------------------------------------------------------
// fp32 -> bf16 elementwise convert
// ---------------------------------------------------------------------------
__global__ void cvt_kernel(const float* __restrict__ x, bf16* __restrict__ y,
                           size_t n) {
  size_t i = (size_t)blockIdx.x * 256 + threadIdx.x;
  if (i < n) y[i] = (bf16)x[i];
}

// ---------------------------------------------------------------------------
// Generic bf16 GEMM:  C(M,N) f32  =  A(M,K) bf16  x  W(N,K)^T bf16
// Block: 256 threads (8 waves), tile 128x256, K-tile 64, double-buffered LDS
// with async global->LDS staging. Wave grid 2x4; each wave computes 64x64
// via 4x4 WMMA 16x16 accumulators (16 WMMA : 16 ds_load_b128 per K-step).
// ---------------------------------------------------------------------------
__global__ __launch_bounds__(256) void gemm_bf16_kernel(
    const bf16* __restrict__ A, const bf16* __restrict__ W,
    float* __restrict__ C, int M, int N, int K) {
  __shared__ __align__(16) bf16 lA[2][128 * 64];   // 2 x 16KB
  __shared__ __align__(16) bf16 lB[2][256 * 64];   // 2 x 32KB

  const int t    = threadIdx.x;
  const int lane = t & 31;
  const int wvid = t >> 5;
  const int wm   = wvid >> 2;        // 0..1
  const int wn   = wvid & 3;         // 0..3
  const int hf   = lane >> 4;        // lane half (CDNA5 wave32 frag layout)
  const int l16  = lane & 15;
  const int m0   = blockIdx.x * 128;
  const int n0   = blockIdx.y * 256;

  v8f acc[4][4] = {};

  auto stage = [&](int buf, int k0) {
#pragma unroll
    for (int i = 0; i < 4; ++i) {                 // A: 1024 x 16B chunks
      int idx = t + i * 256, row = idx >> 3, c = idx & 7;
      cp16_async(&lA[buf][row * 64 + c * 8],
                 &A[(size_t)(m0 + row) * K + k0 + c * 8]);
    }
#pragma unroll
    for (int i = 0; i < 8; ++i) {                 // W: 2048 x 16B chunks
      int idx = t + i * 256, row = idx >> 3, c = idx & 7;
      cp16_async(&lB[buf][row * 64 + c * 8],
                 &W[(size_t)(n0 + row) * K + k0 + c * 8]);
    }
  };

  const int nk = K / 64;
  stage(0, 0);
  wait_async0();
  __syncthreads();

  for (int kt = 0; kt < nk; ++kt) {
    const int buf = kt & 1;
    if (kt + 1 < nk) stage(buf ^ 1, (kt + 1) * 64);   // prefetch next tile

#pragma unroll
    for (int kk = 0; kk < 2; ++kk) {      // two K=32 WMMA steps
      BFrag a[4], b[4];
#pragma unroll
      for (int mt = 0; mt < 4; ++mt) {
        const bf16* p = &lA[buf][(wm * 64 + mt * 16 + l16) * 64 + kk * 32];
        a[mt].h[0] = *(const v8bf*)(p + hf * 8);        // K = h*8 .. +7
        a[mt].h[1] = *(const v8bf*)(p + 16 + hf * 8);   // K = 16+h*8 .. +7
      }
#pragma unroll
      for (int nt = 0; nt < 4; ++nt) {
        const bf16* p = &lB[buf][(wn * 64 + nt * 16 + l16) * 64 + kk * 32];
        b[nt].h[0] = *(const v8bf*)(p + hf * 8);        // K = h*8 .. +7
        b[nt].h[1] = *(const v8bf*)(p + 16 + hf * 8);   // K = 16+h*8 .. +7
      }
#pragma unroll
      for (int mt = 0; mt < 4; ++mt)
#pragma unroll
        for (int nt = 0; nt < 4; ++nt)
          acc[mt][nt] = wmma_bf16(a[mt].v, b[nt].v, acc[mt][nt]);
    }
    wait_async0();        // next tile landed (overlapped with WMMA above)
    __syncthreads();
  }

  // ---- store (C layout: VGPR r -> row half*8+r, lane -> col) ----
#pragma unroll
  for (int mt = 0; mt < 4; ++mt)
#pragma unroll
    for (int nt = 0; nt < 4; ++nt) {
      int col = n0 + wn * 64 + nt * 16 + l16;
#pragma unroll
      for (int r = 0; r < 8; ++r) {
        int row = m0 + wm * 64 + mt * 16 + hf * 8 + r;
        C[(size_t)row * N + col] = acc[mt][nt][r];
      }
    }
}

// ---------------------------------------------------------------------------
// RoPE: read f32 proj (b,s,h,d), write bf16 (b,h,s,d). Thread = one (d,d+128).
// ---------------------------------------------------------------------------
__global__ void rope_kernel(const float* __restrict__ X, bf16* __restrict__ Y,
                            int nheads) {
  size_t i = (size_t)blockIdx.x * 256 + threadIdx.x;   // pair index
  int d = (int)(i & 127);
  size_t tmp = i >> 7;
  int h = (int)(tmp % nheads); tmp /= nheads;
  int s = (int)(tmp % S_);
  int b = (int)(tmp / S_);
  float inv_freq = __expf(-(float)d * (9.210340371976184f / 128.0f)); // 10000^(-d/128)
  float theta = (float)s * inv_freq;
  float c, sn;
  __sincosf(theta, &sn, &c);
  const float* row = X + ((size_t)(b * S_ + s)) * ((size_t)nheads * D_) + (size_t)h * D_;
  float x0 = row[d];
  float x1 = row[d + 128];
  bf16* out = Y + (((size_t)(b * nheads + h)) * S_ + s) * D_;
  out[d]       = (bf16)(x0 * c - x1 * sn);
  out[d + 128] = (bf16)(x1 * c + x0 * sn);
}

// ---------------------------------------------------------------------------
// V transpose: f32 (b,s,hkv,d) -> bf16 (b,hkv,d,s)
// ---------------------------------------------------------------------------
__global__ void vT_kernel(const float* __restrict__ X, bf16* __restrict__ Y) {
  size_t i = (size_t)blockIdx.x * 256 + threadIdx.x;   // over B*S*HKV*D
  int d = (int)(i & 255);
  size_t tmp = i >> 8;
  int h = (int)(tmp % HKV_); tmp /= HKV_;
  int s = (int)(tmp % S_);
  int b = (int)(tmp / S_);
  float v = X[((size_t)(b * S_ + s)) * (HKV_ * D_) + (size_t)h * D_ + d];
  Y[(((size_t)(b * HKV_ + h)) * D_ + d) * S_ + s] = (bf16)v;
}

// ---------------------------------------------------------------------------
// Flash attention (causal, GQA group=2), bf16 WMMA, f32 online softmax.
// Block = 128 threads (4 waves), one (b,h) x 64 q-rows; each wave owns 16 rows.
// K/V tiles double-buffered in LDS with async global->LDS staging.
// Q (b,h,s,d) bf16; K (b,hkv,s,d) bf16; Vt (b,hkv,d,s) bf16.
// Out: bf16 (b, s, h*256+d)  -> rows for the final GEMM.
// ---------------------------------------------------------------------------
__global__ __launch_bounds__(128) void attn_kernel(
    const bf16* __restrict__ Q, const bf16* __restrict__ Kb,
    const bf16* __restrict__ Vt, bf16* __restrict__ Ob) {
  __shared__ __align__(16) bf16 lK[2][64 * 256];   // [kv][d]   2 x 32KB
  __shared__ __align__(16) bf16 lV[2][256 * 64];   // [d][kv]   2 x 32KB
  __shared__ __align__(16) bf16 lP[4][16 * 64];    // per-wave P [m][kv] 8KB

  const int t    = threadIdx.x;
  const int lane = t & 31;
  const int wvid = t >> 5;       // 0..3
  const int hf   = lane >> 4;
  const int l16  = lane & 15;
  const int qb   = blockIdx.x;   // 64-row q tile
  const int bh   = blockIdx.y;   // b*8 + h
  const int b    = bh >> 3, h = bh & 7, hkv = h >> 1;

  const bf16* Qp = Q  + ((size_t)bh * S_ + (size_t)qb * 64 + wvid * 16) * D_;
  const bf16* Kp = Kb + ((size_t)(b * HKV_ + hkv) * S_) * D_;
  const bf16* Vp = Vt + ((size_t)(b * HKV_ + hkv) * D_) * S_;

  // pin Q fragments in registers: 8 K-steps of 32
  BFrag qf[8];
#pragma unroll
  for (int ks = 0; ks < 8; ++ks) {
    const bf16* qr = Qp + (size_t)l16 * D_ + ks * 32;
    qf[ks].h[0] = *(const v8bf*)(qr + hf * 8);
    qf[ks].h[1] = *(const v8bf*)(qr + 16 + hf * 8);
  }

  v8f o[16] = {};                // 16 d-tiles x 16x16 accum
  float mrun[8], lrun[8];
#pragma unroll
  for (int r = 0; r < 8; ++r) { mrun[r] = -INFINITY; lrun[r] = 0.0f; }

  const int qrow_base = qb * 64 + wvid * 16 + hf * 8;
  const int nkv = qb + 1;        // causal: kv tiles 0..qb

  auto stage = [&](int buf, int kt) {
#pragma unroll
    for (int i = 0; i < 16; ++i) {             // K tile: 2048 x 16B chunks
      int idx = t + i * 128;
      int row = idx >> 5, c = idx & 31;        // 32 chunks per 256-elem row
      cp16_async(&lK[buf][row * 256 + c * 8],
                 &Kp[(size_t)(kt * 64 + row) * D_ + c * 8]);
    }
#pragma unroll
    for (int i = 0; i < 16; ++i) {             // V^T tile: 2048 x 16B chunks
      int idx = t + i * 128;
      int row = idx >> 3, c = idx & 7;         // 8 chunks per 64-elem row
      cp16_async(&lV[buf][row * 64 + c * 8],
                 &Vp[(size_t)row * S_ + (size_t)kt * 64 + c * 8]);
    }
  };

  stage(0, 0);
  wait_async0();
  __syncthreads();

  for (int kt = 0; kt < nkv; ++kt) {
    const int buf = kt & 1;
    if (kt + 1 < nkv) stage(buf ^ 1, kt + 1);  // prefetch next kv tile

    // ---- S = Q K^T over this tile: 4 col-tiles x 8 K-steps ----
    v8f sa[4] = {};
#pragma unroll
    for (int nt = 0; nt < 4; ++nt) {
#pragma unroll
      for (int ks = 0; ks < 8; ++ks) {
        BFrag kb;
        const bf16* p = &lK[buf][(nt * 16 + l16) * 256 + ks * 32];
        kb.h[0] = *(const v8bf*)(p + hf * 8);
        kb.h[1] = *(const v8bf*)(p + 16 + hf * 8);
        sa[nt] = wmma_bf16(qf[ks].v, kb.v, sa[nt]);
      }
    }

    // ---- scale + causal mask ----
    const float scale = 0.0625f;             // 1/sqrt(256)
#pragma unroll
    for (int nt = 0; nt < 4; ++nt) {
      int ng = kt * 64 + nt * 16 + l16;
#pragma unroll
      for (int r = 0; r < 8; ++r) {
        int mg = qrow_base + r;
        sa[nt][r] = sa[nt][r] * scale + ((ng <= mg) ? 0.0f : -1.0e9f);
      }
    }

    // ---- online softmax: row stats via 16-lane shfl reductions ----
    float mnew[8];
#pragma unroll
    for (int r = 0; r < 8; ++r) {
      float v = fmaxf(fmaxf(sa[0][r], sa[1][r]), fmaxf(sa[2][r], sa[3][r]));
#pragma unroll
      for (int off = 1; off < 16; off <<= 1) v = fmaxf(v, __shfl_xor(v, off, 16));
      mnew[r] = fmaxf(mrun[r], v);
    }
    float rs[8];
#pragma unroll
    for (int r = 0; r < 8; ++r) rs[r] = 0.0f;
#pragma unroll
    for (int nt = 0; nt < 4; ++nt)
#pragma unroll
      for (int r = 0; r < 8; ++r) {
        float p = __expf(sa[nt][r] - mnew[r]);
        sa[nt][r] = p;
        rs[r] += p;
      }
#pragma unroll
    for (int r = 0; r < 8; ++r) {
#pragma unroll
      for (int off = 1; off < 16; off <<= 1) rs[r] += __shfl_xor(rs[r], off, 16);
    }
    float alpha[8];
#pragma unroll
    for (int r = 0; r < 8; ++r) {
      alpha[r] = __expf(mrun[r] - mnew[r]);
      lrun[r]  = lrun[r] * alpha[r] + rs[r];
      mrun[r]  = mnew[r];
    }
#pragma unroll
    for (int dt = 0; dt < 16; ++dt)
#pragma unroll
      for (int r = 0; r < 8; ++r) o[dt][r] *= alpha[r];

    // ---- P -> per-wave LDS (C-layout -> A-layout relay) ----
#pragma unroll
    for (int nt = 0; nt < 4; ++nt)
#pragma unroll
      for (int r = 0; r < 8; ++r)
        lP[wvid][(hf * 8 + r) * 64 + nt * 16 + l16] = (bf16)sa[nt][r];
    // same-wave LDS ops are in order: no cross-wave barrier needed for lP

    // ---- O += P V : 2 K-steps of 32 x 16 d-tiles ----
#pragma unroll
    for (int ks2 = 0; ks2 < 2; ++ks2) {
      BFrag pa;
      const bf16* pp = &lP[wvid][l16 * 64 + ks2 * 32];
      pa.h[0] = *(const v8bf*)(pp + hf * 8);
      pa.h[1] = *(const v8bf*)(pp + 16 + hf * 8);
#pragma unroll
      for (int dt = 0; dt < 16; ++dt) {
        BFrag vb;
        const bf16* p = &lV[buf][(dt * 16 + l16) * 64 + ks2 * 32];
        vb.h[0] = *(const v8bf*)(p + hf * 8);
        vb.h[1] = *(const v8bf*)(p + 16 + hf * 8);
        o[dt] = wmma_bf16(pa.v, vb.v, o[dt]);
      }
    }
    wait_async0();      // prefetched tile landed (overlapped with compute)
    __syncthreads();    // protect lK/lV buffers
  }

  // ---- normalize and store bf16 (b, s, h*256+d) ----
  float inv[8];
#pragma unroll
  for (int r = 0; r < 8; ++r) inv[r] = 1.0f / lrun[r];
#pragma unroll
  for (int dt = 0; dt < 16; ++dt)
#pragma unroll
    for (int r = 0; r < 8; ++r) {
      int srow = qb * 64 + wvid * 16 + hf * 8 + r;
      Ob[((size_t)(b * S_ + srow)) * (H_ * D_) + h * D_ + dt * 16 + l16] =
          (bf16)(o[dt][r] * inv[r]);
    }
}

// ---------------------------------------------------------------------------
// host-side launch
// ---------------------------------------------------------------------------
extern "C" void kernel_launch(void* const* d_in, const int* in_sizes, int n_in,
                              void* d_out, int out_size, void* d_ws, size_t ws_size,
                              hipStream_t stream) {
  (void)in_sizes; (void)n_in; (void)out_size; (void)ws_size;
  const float* hid = (const float*)d_in[0];
  // d_in[1] attention_mask: causal + -1e9, recomputed in-kernel
  // d_in[2] position_ids:   arange, recomputed in-kernel
  const float* Wq = (const float*)d_in[3];
  const float* Wk = (const float*)d_in[4];
  const float* Wv = (const float*)d_in[5];
  const float* Wo = (const float*)d_in[6];
  float* out = (float*)d_out;

  char* ws = (char*)d_ws;
  size_t off = 0;
  auto alloc = [&](size_t bytes) -> void* {
    void* p = ws + off;
    off = (off + bytes + 255) & ~(size_t)255;
    return p;
  };
  bf16*  hid_bf = (bf16*)alloc((size_t)MTOT * IN_ * 2);
  bf16*  wq_bf  = (bf16*)alloc((size_t)H_ * D_ * IN_ * 2);
  bf16*  wk_bf  = (bf16*)alloc((size_t)HKV_ * D_ * IN_ * 2);
  bf16*  wv_bf  = (bf16*)alloc((size_t)HKV_ * D_ * IN_ * 2);
  bf16*  wo_bf  = (bf16*)alloc((size_t)HID_ * H_ * D_ * 2);
  float* qf     = (float*)alloc((size_t)MTOT * H_ * D_ * 4);
  float* kf     = (float*)alloc((size_t)MTOT * HKV_ * D_ * 4);
  float* vf     = (float*)alloc((size_t)MTOT * HKV_ * D_ * 4);
  bf16*  q_bf   = (bf16*)alloc((size_t)MTOT * H_ * D_ * 2);
  bf16*  k_bf   = (bf16*)alloc((size_t)MTOT * HKV_ * D_ * 2);
  bf16*  v_bf   = (bf16*)alloc((size_t)MTOT * HKV_ * D_ * 2);
  bf16*  att_bf = (bf16*)qf;   // alias: qf is dead after rope_q

  auto cvt = [&](const float* x, bf16* y, size_t n) {
    cvt_kernel<<<dim3((unsigned)((n + 255) / 256)), 256, 0, stream>>>(x, y, n);
  };
  cvt(hid, hid_bf, (size_t)MTOT * IN_);
  cvt(Wq, wq_bf, (size_t)H_ * D_ * IN_);
  cvt(Wk, wk_bf, (size_t)HKV_ * D_ * IN_);
  cvt(Wv, wv_bf, (size_t)HKV_ * D_ * IN_);
  cvt(Wo, wo_bf, (size_t)HID_ * H_ * D_);

  // projections (block tile 128x256)
  gemm_bf16_kernel<<<dim3(MTOT / 128, (H_ * D_) / 256), 256, 0, stream>>>(
      hid_bf, wq_bf, qf, MTOT, H_ * D_, IN_);
  gemm_bf16_kernel<<<dim3(MTOT / 128, (HKV_ * D_) / 256), 256, 0, stream>>>(
      hid_bf, wk_bf, kf, MTOT, HKV_ * D_, IN_);
  gemm_bf16_kernel<<<dim3(MTOT / 128, (HKV_ * D_) / 256), 256, 0, stream>>>(
      hid_bf, wv_bf, vf, MTOT, HKV_ * D_, IN_);

  // RoPE + V transpose, to bf16 attention layouts
  rope_kernel<<<dim3((B_ * S_ * H_ * 128) / 256), 256, 0, stream>>>(qf, q_bf, H_);
  rope_kernel<<<dim3((B_ * S_ * HKV_ * 128) / 256), 256, 0, stream>>>(kf, k_bf, HKV_);
  vT_kernel<<<dim3((B_ * S_ * HKV_ * D_) / 256), 256, 0, stream>>>(vf, v_bf);

  // attention
  attn_kernel<<<dim3(S_ / 64, B_ * H_), 128, 0, stream>>>(q_bf, k_bf, v_bf, att_bf);

  // output projection -> d_out
  gemm_bf16_kernel<<<dim3(MTOT / 128, HID_ / 256), 256, 0, stream>>>(
      att_bf, wo_bf, out, MTOT, HID_, H_ * D_);
}
